// PIPNet_90452011254231
// MI455X (gfx1250) — matrix operand order repaired and checked by reading, not compiled, not run on hardware
//
#include <hip/hip_runtime.h>
#include <hip/hip_bf16.h>

typedef __attribute__((ext_vector_type(16))) _Float16 v16h;
typedef __attribute__((ext_vector_type(8)))  float    v8f;

#define P_PAIRS 4096
#define N_VERTS 40000
#define C_FEAT  64
#define NCHUNK  1250          // 40000 / 32
#define WAVES   4

// ---------------------------------------------------------------------------
// Kernel 1: pack feats (f32, N x 64 row-major) into the exact 16-bit B-matrix
// (32x16) VGPR/lane layout this file feeds to v_wmma_f32_16x16x32_f16, pre-converted to
// f16.  One thread produces one lane's 16 halfs (32 contiguous bytes) for one
// (chunk, n-tile) pair, so the main kernel loads B with contiguous b128s.
//
// B layout used here (contiguous K): lane l holds column N = l&15; lanes 0-15 hold
// K=0..15 (2 per VGPR), lanes 16-31 hold K=16..31.
// ---------------------------------------------------------------------------
__global__ __launch_bounds__(256) void pack_feats_kernel(
    const float* __restrict__ feats, _Float16* __restrict__ out)
{
    int t = blockIdx.x * blockDim.x + threadIdx.x;   // t = (c*4 + j)*32 + lane
    if (t >= NCHUNK * 4 * 32) return;
    int lane = t & 31;
    int j    = (t >> 5) & 3;          // n-tile (feature columns 16j..16j+15)
    int c    = t >> 7;                // k-chunk
    int kbase = c * 32 + ((lane >> 4) * 16);
    int col   = j * 16 + (lane & 15);
    v16h v;
#pragma unroll
    for (int e = 0; e < 16; ++e)
        v[e] = (_Float16)feats[(size_t)(kbase + e) * C_FEAT + col];
    *(v16h*)(out + (size_t)t * 16) = v;
}

// ---------------------------------------------------------------------------
// Kernel 2: fused RBF + GEMM.  Block = 16 pairs (one M tile), 4 waves split
// the 1250 K-chunks.  A-matrix (16x32 f16 weights) is generated in-register:
// lane = m + 16*g holds pair m and K-set g (contiguous K order, as in the B pack:
// group 0 -> K = 0..15 ; group 1 -> K = 16..31).
// A 5th WMMA against an all-ones B matrix produces the row sums (norm), so
// numerator and denominator use identical f16 weights (like the reference).
// Outputs: F[p*64+c] = sum_k w(p,k)*feats(k,c)  (unnormalized), S[p] = sum_k w.
// ---------------------------------------------------------------------------
__global__ __launch_bounds__(128) void rbf_wmma_kernel(
    const float* __restrict__ pairs_loc, int side,
    const float* __restrict__ verts,
    const _Float16* __restrict__ bpack,
    float* __restrict__ F, float* __restrict__ S)
{
    const int lane = threadIdx.x & 31;
    const int wave = threadIdx.x >> 5;
    const int m    = lane & 15;
    const int p    = blockIdx.x * 16 + m;

    const float* pl = pairs_loc + ((size_t)p * 2 + side) * 3;
    const float px = pl[0], py = pl[1], pz = pl[2];

    v8f acc0 = {}, acc1 = {}, acc2 = {}, acc3 = {}, accS = {};
    const float C2 = -0.57707801635558534f;        // -1/(2.5 * ln2)
    const int srcbase = (lane >> 4) * 16;

    v16h ones;
#pragma unroll
    for (int e = 0; e < 16; ++e) ones[e] = (_Float16)1.0f;

    // prefetch first vertex triple (coalesced b96)
    const float* vp0 = verts + (size_t)(wave * 32 + lane) * 3;
    float nvx = vp0[0], nvy = vp0[1], nvz = vp0[2];

    for (int c = wave; c < NCHUNK; c += WAVES) {
        // issue B loads first so their latency is hidden by A generation
        const _Float16* bb = bpack + (((size_t)c * 4) * 32 + lane) * 16;
        const v16h b0 = *(const v16h*)(bb);
        const v16h b1 = *(const v16h*)(bb + 32 * 16);
        const v16h b2 = *(const v16h*)(bb + 2 * 32 * 16);
        const v16h b3 = *(const v16h*)(bb + 3 * 32 * 16);

        const float vx = nvx, vy = nvy, vz = nvz;
        // prefetch next chunk's vertex (clamped address, branch-free)
        int cn = c + WAVES; cn = cn < NCHUNK ? cn : (NCHUNK - 1);
        const float* vq = verts + (size_t)(cn * 32 + lane) * 3;
        nvx = vq[0]; nvy = vq[1]; nvz = vq[2];

        v16h a;
#pragma unroll
        for (int kk = 0; kk < 16; ++kk) {
            // source lane holding vertex K = srcbase + kk for this lane group
            const int src = kk | srcbase;
            const float sx = __shfl(vx, src, 32);
            const float sy = __shfl(vy, src, 32);
            const float sz = __shfl(vz, src, 32);
            const float dx = px - sx, dy = py - sy, dz = pz - sz;
            const float d2 = dx * dx + dy * dy + dz * dz;
            const float d  = __builtin_amdgcn_sqrtf(d2);
            const float w  = __builtin_amdgcn_exp2f(d * C2);   // exp(-d/2.5)
            a[kk] = (_Float16)w;
        }

        acc0 = __builtin_amdgcn_wmma_f32_16x16x32_f16(false, a, false, b0,
                                                      (short)0, acc0, false, false);
        acc1 = __builtin_amdgcn_wmma_f32_16x16x32_f16(false, a, false, b1,
                                                      (short)0, acc1, false, false);
        acc2 = __builtin_amdgcn_wmma_f32_16x16x32_f16(false, a, false, b2,
                                                      (short)0, acc2, false, false);
        acc3 = __builtin_amdgcn_wmma_f32_16x16x32_f16(false, a, false, b3,
                                                      (short)0, acc3, false, false);
        accS = __builtin_amdgcn_wmma_f32_16x16x32_f16(false, a, false, ones,
                                                      (short)0, accS, false, false);
    }

    // ------- cross-wave reduction through LDS -------
    // per-wave region: 4*256 floats of F tiles + 256 floats of the sum tile
    __shared__ float red[WAVES * 1280];
#pragma unroll
    for (int r = 0; r < 8; ++r) {
        red[wave * 1280 + 0 * 256 + r * 32 + lane] = acc0[r];
        red[wave * 1280 + 1 * 256 + r * 32 + lane] = acc1[r];
        red[wave * 1280 + 2 * 256 + r * 32 + lane] = acc2[r];
        red[wave * 1280 + 3 * 256 + r * 32 + lane] = acc3[r];
        red[wave * 1280 + 4 * 256 + r * 32 + lane] = accS[r];
    }
    __syncthreads();

    for (int f = threadIdx.x; f < 1024; f += 128) {
        float s = red[f] + red[1280 + f] + red[2560 + f] + red[3840 + f];
        const int j = f >> 8, r = (f >> 5) & 7, l = f & 31;
        const int M   = r + 8 * (l >> 4);            // C/D layout row
        const int col = j * 16 + (l & 15);           // C/D layout col
        F[(size_t)(blockIdx.x * 16 + M) * C_FEAT + col] = s;
    }
    if (threadIdx.x < 16) {
        // row M lives at (r = M&7, lane = 16*(M>>3) + n); take column n = 0
        const int M = threadIdx.x;
        const int off = 1024 + (M & 7) * 32 + ((M >> 3) * 16);
        float s = 0.f;
#pragma unroll
        for (int w = 0; w < WAVES; ++w) s += red[w * 1280 + off];
        S[blockIdx.x * 16 + M] = s;
    }
}

// ---------------------------------------------------------------------------
// Kernel 3: tiny MLP  x(130) -> relu(128) -> relu(128) -> 1.
// One block per pair; W matrices are L2-resident (130KB).
// ---------------------------------------------------------------------------
__global__ __launch_bounds__(128) void mlp_kernel(
    const float* __restrict__ FL, const float* __restrict__ SL,
    const float* __restrict__ FR, const float* __restrict__ SR,
    const float* __restrict__ W1, const float* __restrict__ b1,
    const float* __restrict__ W2, const float* __restrict__ b2,
    const float* __restrict__ W3, const float* __restrict__ b3,
    float* __restrict__ out)
{
    const int p = blockIdx.x;
    const int t = threadIdx.x;
    __shared__ float x[130];
    __shared__ float h[128];

    const float nl = SL[p] + 0.01f;
    const float nr = SR[p] + 0.01f;
    if (t < 64)       x[t]     = FL[(size_t)p * 64 + t] / nl;
    else              x[t + 1] = FR[(size_t)p * 64 + (t - 64)] / nr;
    if (t == 0) x[64]  = tanhf(nl);
    if (t == 1) x[129] = tanhf(nr);
    __syncthreads();

    float s = b1[t];
    for (int i = 0; i < 130; ++i) s += x[i] * W1[i * 128 + t];
    s = fmaxf(s, 0.f);
    h[t] = s;
    __syncthreads();

    float s2 = b2[t];
    for (int i = 0; i < 128; ++i) s2 += h[i] * W2[i * 128 + t];
    s2 = fmaxf(s2, 0.f);
    const float contrib = s2 * W3[t];
    __syncthreads();
    h[t] = contrib;
    __syncthreads();
    for (int off = 64; off > 0; off >>= 1) {
        if (t < off) h[t] += h[t + off];
        __syncthreads();
    }
    if (t == 0) out[p] = h[0] + b3[0];
}

// ---------------------------------------------------------------------------
extern "C" void kernel_launch(void* const* d_in, const int* in_sizes, int n_in,
                              void* d_out, int out_size, void* d_ws, size_t ws_size,
                              hipStream_t stream)
{
    const float* pairs_loc = (const float*)d_in[0];
    const float* verts_l   = (const float*)d_in[1];
    const float* verts_r   = (const float*)d_in[2];
    const float* proc_l    = (const float*)d_in[3];
    const float* proc_r    = (const float*)d_in[4];
    const float* W1 = (const float*)d_in[5];
    const float* b1 = (const float*)d_in[6];
    const float* W2 = (const float*)d_in[7];
    const float* b2 = (const float*)d_in[8];
    const float* W3 = (const float*)d_in[9];
    const float* b3 = (const float*)d_in[10];

    char* ws = (char*)d_ws;
    const size_t PACK_BYTES = (size_t)NCHUNK * 4 * 32 * 16 * sizeof(_Float16); // 5.12 MB
    const size_t F_BYTES    = (size_t)P_PAIRS * C_FEAT * sizeof(float);        // 1 MB
    const size_t S_BYTES    = (size_t)P_PAIRS * sizeof(float);

    _Float16* packL = (_Float16*)(ws);
    _Float16* packR = (_Float16*)(ws + PACK_BYTES);
    float* FL = (float*)(ws + 2 * PACK_BYTES);
    float* FR = (float*)(ws + 2 * PACK_BYTES + F_BYTES);
    float* SL = (float*)(ws + 2 * PACK_BYTES + 2 * F_BYTES);
    float* SR = (float*)(ws + 2 * PACK_BYTES + 2 * F_BYTES + S_BYTES);

    const int packThreads = NCHUNK * 4 * 32;               // 160000
    pack_feats_kernel<<<(packThreads + 255) / 256, 256, 0, stream>>>(proc_l, packL);
    pack_feats_kernel<<<(packThreads + 255) / 256, 256, 0, stream>>>(proc_r, packR);

    rbf_wmma_kernel<<<P_PAIRS / 16, 128, 0, stream>>>(pairs_loc, 0, verts_l, packL, FL, SL);
    rbf_wmma_kernel<<<P_PAIRS / 16, 128, 0, stream>>>(pairs_loc, 1, verts_r, packR, FR, SR);

    mlp_kernel<<<P_PAIRS, 128, 0, stream>>>(FL, SL, FR, SR,
                                            W1, b1, W2, b2, W3, b3,
                                            (float*)d_out);
}
